// LSTMCell_25829933318237
// MI455X (gfx1250) — matrix operand order repaired and checked by reading, compiled, not run on hardware
//
#include <hip/hip_runtime.h>
#include <hip/hip_bf16.h>

#define B_DIM  8192
#define N_INP  1024
#define N_OUT  1024
#define K_DIM  2048   // N_INP + N_OUT

typedef __attribute__((ext_vector_type(16))) __bf16 v16bf;
typedef __attribute__((ext_vector_type(8)))  __bf16 v8bf;
typedef __attribute__((ext_vector_type(8)))  float  v8f;

// fp32 -> bf16 round-to-nearest-even
static __device__ __forceinline__ unsigned short f2bf(float f) {
  unsigned int u = __float_as_uint(f);
  u += 0x7FFFu + ((u >> 16) & 1u);
  return (unsigned short)(u >> 16);
}

static __device__ __forceinline__ float sigmoid_f(float x) {
  return 1.0f / (1.0f + __expf(-x));
}
static __device__ __forceinline__ float tanh_f(float x) {
  x = fminf(fmaxf(x, -15.0f), 15.0f);
  float e = __expf(2.0f * x);
  return (e - 1.0f) / (e + 1.0f);
}

// ---------------------------------------------------------------------------
// Kernel 1: xh = concat(x, out_tm1) converted to bf16, row-major [B, 2048]
// ---------------------------------------------------------------------------
__global__ void pack_xh_bf16(const float* __restrict__ x,
                             const float* __restrict__ h,
                             unsigned short* __restrict__ xh) {
  size_t idx = (size_t)blockIdx.x * blockDim.x + threadIdx.x;  // 8 elems/thread
  size_t row = idx >> 8;                    // 2048/8 = 256 chunks per row
  int    k0  = ((int)idx & 255) << 3;
  const float* src = (k0 < N_INP) ? (x + row * N_INP + k0)
                                  : (h + row * N_OUT + (k0 - N_INP));
  float4 f0 = reinterpret_cast<const float4*>(src)[0];
  float4 f1 = reinterpret_cast<const float4*>(src)[1];
  uint4 o;
  o.x = (unsigned)f2bf(f0.x) | ((unsigned)f2bf(f0.y) << 16);
  o.y = (unsigned)f2bf(f0.z) | ((unsigned)f2bf(f0.w) << 16);
  o.z = (unsigned)f2bf(f1.x) | ((unsigned)f2bf(f1.y) << 16);
  o.w = (unsigned)f2bf(f1.z) | ((unsigned)f2bf(f1.w) << 16);
  reinterpret_cast<uint4*>(xh + row * K_DIM + k0)[0] = o;
}

// ---------------------------------------------------------------------------
// Kernel 2: Wt[g][n][k] = (bf16) W_g[k][n]   (transposed weights, [N,K] rows)
// ---------------------------------------------------------------------------
struct WPtrs { const float* w[4]; };

__global__ void transpose_w_bf16(WPtrs wp, unsigned short* __restrict__ wt) {
  __shared__ unsigned short tile[32][33];
  int g  = blockIdx.z;
  int kt = blockIdx.y;   // 64 tiles along K
  int nt = blockIdx.x;   // 32 tiles along N
  const float* w = wp.w[g];
  int c = threadIdx.x & 31;
  int r = threadIdx.x >> 5;          // 0..7
#pragma unroll
  for (int i = 0; i < 4; ++i) {
    int rr = r + i * 8;
    tile[rr][c] = f2bf(w[(size_t)(kt * 32 + rr) * N_OUT + nt * 32 + c]);
  }
  __syncthreads();
  unsigned short* dst = wt + (size_t)g * N_OUT * K_DIM;
#pragma unroll
  for (int i = 0; i < 4; ++i) {
    int rr = r + i * 8;
    dst[(size_t)(nt * 32 + rr) * K_DIM + kt * 32 + c] = tile[c][rr];
  }
}

// ---------------------------------------------------------------------------
// Kernel 3: fused 4-gate GEMM (bf16 WMMA, f32 acc) + LSTM pointwise epilogue
// Workgroup: 8 waves; wave grid 4(M) x 2(N); wave tile 32x16 per gate.
// Block tile: 128 (M) x 32 (N). Grid: (1024/32, 8192/128).
// ---------------------------------------------------------------------------
static __device__ __forceinline__ v16bf ld_frag_a(const __bf16* p) {
  // per-lane: 8 contiguous bf16 at p, 8 contiguous at p+16 (ISA A layout)
  v8bf l = *reinterpret_cast<const v8bf*>(p);
  v8bf h = *reinterpret_cast<const v8bf*>(p + 16);
  return __builtin_shufflevector(l, h, 0,1,2,3,4,5,6,7,8,9,10,11,12,13,14,15);
}
static __device__ __forceinline__ v16bf ld_frag_b(const __bf16* p) {
  // per-lane: 8 contiguous bf16 at p, 8 contiguous at p+16 of one weight row
  v8bf l = *reinterpret_cast<const v8bf*>(p);
  v8bf h = *reinterpret_cast<const v8bf*>(p + 16);
  return __builtin_shufflevector(l, h, 0,1,2,3,4,5,6,7,8,9,10,11,12,13,14,15);
}

__global__ void __launch_bounds__(256, 2)
lstm_gemm_fused(const unsigned short* __restrict__ xh_u16,
                const unsigned short* __restrict__ wt_u16,
                const float* __restrict__ state_tm1,
                const float* __restrict__ bfv, const float* __restrict__ biv,
                const float* __restrict__ bcv, const float* __restrict__ bov,
                float* __restrict__ out) {
  const __bf16* xh = reinterpret_cast<const __bf16*>(xh_u16);
  const __bf16* wt = reinterpret_cast<const __bf16*>(wt_u16);

  const int lane = threadIdx.x & 31;
  const int wave = threadIdx.x >> 5;
  const int wm = wave & 3;            // 0..3 along M
  const int wn = wave >> 2;           // 0..1 along N
  const int lm = lane & 15;
  const bool lo = lane < 16;

  const int m_base = blockIdx.y * 128 + wm * 32;
  const int n_base = blockIdx.x * 32 + wn * 16;
  const int col    = n_base + lm;     // output column for this lane

  // A: lanes 0-15 hold K {0..7,16..23}, lanes 16-31 hold K {8..15,24..31}
  const __bf16* aptr0 = xh + (size_t)(m_base + lm) * K_DIM + (lo ? 0 : 8);
  const __bf16* aptr1 = aptr0 + (size_t)16 * K_DIM;
  // B: lanes 0-15 hold K {0..7,16..23} of col, lanes 16-31 hold K {8..15,24..31}
  const size_t gstride = (size_t)N_OUT * K_DIM;
  const __bf16* bptr = wt + (size_t)col * K_DIM + (lo ? 0 : 8);

  v8f acc[4][2];
#pragma unroll
  for (int g = 0; g < 4; ++g) {
    acc[g][0] = (v8f)(0.0f);
    acc[g][1] = (v8f)(0.0f);
  }

  for (int k = 0; k < K_DIM; k += 32) {
    v16bf a0 = ld_frag_a(aptr0 + k);
    v16bf a1 = ld_frag_a(aptr1 + k);
#pragma unroll
    for (int g = 0; g < 4; ++g) {
      v16bf b = ld_frag_b(bptr + g * gstride + k);
      acc[g][0] = __builtin_amdgcn_wmma_f32_16x16x32_bf16(
          false, a0, false, b, (short)0, acc[g][0], false, false);
      acc[g][1] = __builtin_amdgcn_wmma_f32_16x16x32_bf16(
          false, a1, false, b, (short)0, acc[g][1], false, false);
    }
  }

  // Epilogue: bias + activations + state update, straight from C/D layout.
  const float bF = bfv[col], bI = biv[col], bC = bcv[col], bO = bov[col];
  float* out_t = out;
  float* st_t  = out + (size_t)B_DIM * N_OUT;

#pragma unroll
  for (int h2 = 0; h2 < 2; ++h2) {
    const int mrow = m_base + h2 * 16;
#pragma unroll
    for (int j = 0; j < 8; ++j) {
      const int row = mrow + (lo ? j : j + 8);   // C/D VGPR j -> row j / j+8
      const size_t off = (size_t)row * N_OUT + col;
      float fg = sigmoid_f(acc[0][h2][j] + bF);
      float ig = sigmoid_f(acc[1][h2][j] + bI);
      float cg = tanh_f   (acc[2][h2][j] + bC);
      float og = sigmoid_f(acc[3][h2][j] + bO);
      float ns = state_tm1[off] * fg + ig * cg;
      st_t[off]  = ns;
      out_t[off] = og * tanh_f(ns);
    }
  }
}

// ---------------------------------------------------------------------------
extern "C" void kernel_launch(void* const* d_in, const int* in_sizes, int n_in,
                              void* d_out, int out_size, void* d_ws, size_t ws_size,
                              hipStream_t stream) {
  const float* x     = (const float*)d_in[0];
  const float* h     = (const float*)d_in[1];
  const float* state = (const float*)d_in[2];
  const float* Wf    = (const float*)d_in[3];
  const float* bf_   = (const float*)d_in[4];
  const float* Wi    = (const float*)d_in[5];
  const float* bi_   = (const float*)d_in[6];
  const float* Wc    = (const float*)d_in[7];
  const float* bc_   = (const float*)d_in[8];
  const float* Wo    = (const float*)d_in[9];
  const float* bo_   = (const float*)d_in[10];

  unsigned short* xh = (unsigned short*)d_ws;               // 32 MB
  unsigned short* wt = xh + (size_t)B_DIM * K_DIM;          // +16 MB

  pack_xh_bf16<<<(B_DIM * K_DIM / 8) / 256, 256, 0, stream>>>(x, h, xh);

  WPtrs wp; wp.w[0] = Wf; wp.w[1] = Wi; wp.w[2] = Wc; wp.w[3] = Wo;
  transpose_w_bf16<<<dim3(32, 64, 4), 256, 0, stream>>>(wp, wt);

  lstm_gemm_fused<<<dim3(N_OUT / 32, B_DIM / 128), 256, 0, stream>>>(
      xh, wt, state, bf_, bi_, bc_, bo_, (float*)d_out);
}
